// Head_3839700762801
// MI455X (gfx1250) — matrix operand order repaired and checked by reading, not compiled, not run on hardware
//
#include <hip/hip_runtime.h>

// Shapes: B=8, T=2048, C=1024, H=64
#define BDIM 8
#define TDIM 2048
#define CDIM 1024
#define HDIM 64

typedef __attribute__((ext_vector_type(16))) _Float16 v16h;
typedef __attribute__((ext_vector_type(8)))  _Float16 v8h;
typedef __attribute__((ext_vector_type(8)))  float    v8f;
typedef int v4i __attribute__((vector_size(16)));

// ---- CDNA5 async global->LDS path (guarded; fallback keeps compiling) ------
#ifdef __has_builtin
#if __has_builtin(__builtin_amdgcn_global_load_async_to_lds_b128)
#define HAVE_ASYNC_LDS 1
#endif
#if __has_builtin(__builtin_amdgcn_s_wait_asynccnt)
#define WAIT_ASYNC(n) __builtin_amdgcn_s_wait_asynccnt(n)
#endif
#endif
#ifndef WAIT_ASYNC
#define WAIT_ASYNC(n) asm volatile("s_wait_asynccnt %0" ::"n"(n) : "memory")
#endif

__device__ __forceinline__ unsigned pkh2(float a, float b) {
  union { _Float16 h[2]; unsigned u; } z;
  z.h[0] = (_Float16)a; z.h[1] = (_Float16)b;
  return z.u;
}

// ---------------------------------------------------------------------------
// Kernel 0: W pre-pass. Wt[z][out][in] f16 (col-major) so projection B-frags
// are single contiguous 32B loads. grid (16,3), block 256, 16 elems/thread.
// ---------------------------------------------------------------------------
__global__ void __launch_bounds__(256)
wt_kernel(const float* __restrict__ Wq, const float* __restrict__ Wk,
          const float* __restrict__ Wv, _Float16* __restrict__ Wt) {
  const int z = blockIdx.y;
  const float* W = (z == 0) ? Wq : ((z == 1) ? Wk : Wv);
  const int base = (blockIdx.x * 256 + threadIdx.x) * 16; // 0..65520
  const int col = base >> 10;          // output dim 0..63
  const int k0  = base & (CDIM - 1);   // input dim
  _Float16* dst = Wt + (size_t)z * HDIM * CDIM + base;
#pragma unroll
  for (int i = 0; i < 16; ++i) dst[i] = (_Float16)W[(size_t)(k0 + i) * HDIM + col];
}

// ---------------------------------------------------------------------------
// Kernel 1: fused q/k/v projection. One block = 128 rows of x, ALL 3 outputs
// (x read once). Wave w owns rows [16w,16w+16). Double-buffered async
// global->LDS copy of the x chunk (per-wave disjoint slices -> NO barriers).
// 12 WMMAs per K=32 chunk per wave.
// ---------------------------------------------------------------------------
__global__ void __launch_bounds__(256)
proj_qkv_kernel(const float* __restrict__ x,
                const _Float16* __restrict__ Wt,
                _Float16* __restrict__ qf,
                _Float16* __restrict__ kf,
                _Float16* __restrict__ vt) {
  const int tid  = threadIdx.x;
  const int w    = tid >> 5;
  const int lane = tid & 31;
  const int n    = lane & 15;
  const int hi   = lane >> 4;
  const int M0   = blockIdx.x * 128;

#ifdef HAVE_ASYNC_LDS
  // 2 bufs x 8 waves x (16 rows x 32 f32) = 32 KB
  __shared__ alignas(16) float Xbuf[2][8][512];
  float* xnc = const_cast<float*>(x);
  // issue 4 async B128 copies (2KB = this wave's 16x32 f32 slice of chunk kc)
  auto issue = [&](int kc, int buf) {
#pragma unroll
    for (int i = 0; i < 4; ++i) {
      const int s = i * 32 + lane;                     // 16B segment id
      float* gsrc = xnc + (size_t)(M0 + 16 * w + (s >> 3)) * CDIM +
                    kc * 32 + (s & 7) * 4;
      __builtin_amdgcn_global_load_async_to_lds_b128(
          (__attribute__((address_space(1))) v4i*)gsrc,
          (__attribute__((address_space(3))) v4i*)(&Xbuf[buf][w][s * 4]),
          0, 0);
    }
  };
  issue(0, 0);
#endif

  v8f acc[3][4] = {};

  for (int kc = 0; kc < CDIM / 32; ++kc) {
    const int k0 = kc * 32;
#ifdef HAVE_ASYNC_LDS
    const int buf = kc & 1;
    if (kc + 1 < CDIM / 32) { issue(kc + 1, buf ^ 1); WAIT_ASYNC(4); }
    else                    { WAIT_ASYNC(0); }
    const float* ap = &Xbuf[buf][w][n * 32];
#else
    const float* ap = x + (size_t)(M0 + 16 * w + n) * CDIM + k0;
#endif
    // A fragment 16x32: lane row = 16w+n, K(h) = h + 16*hi
    float4 f0 = *(const float4*)(ap + 16 * hi);
    float4 f1 = *(const float4*)(ap + 16 * hi + 4);
    float4 f2 = *(const float4*)(ap + 16 * hi + 8);
    float4 f3 = *(const float4*)(ap + 16 * hi + 12);
    v16h a;
    a[0]  = (_Float16)f0.x; a[1]  = (_Float16)f0.y;
    a[2]  = (_Float16)f0.z; a[3]  = (_Float16)f0.w;
    a[4]  = (_Float16)f1.x; a[5]  = (_Float16)f1.y;
    a[6]  = (_Float16)f1.z; a[7]  = (_Float16)f1.w;
    a[8]  = (_Float16)f2.x; a[9]  = (_Float16)f2.y;
    a[10] = (_Float16)f2.z; a[11] = (_Float16)f2.w;
    a[12] = (_Float16)f3.x; a[13] = (_Float16)f3.y;
    a[14] = (_Float16)f3.z; a[15] = (_Float16)f3.w;

#pragma unroll
    for (int z = 0; z < 3; ++z)
#pragma unroll
      for (int j = 0; j < 4; ++j) {
        // B fragment 32x16: lane col = 16j+n, K(h)=h+16hi; Wt row contiguous
        v16h bf = *(const v16h*)(Wt + ((size_t)(z * HDIM + 16 * j + n)) * CDIM +
                                 k0 + 16 * hi);
        acc[z][j] = __builtin_amdgcn_wmma_f32_16x16x32_f16(
            false, a, false, bf, (short)0, acc[z][j], false, false);
      }
  }

  // epilogue (C/D layout: lane col = n, rows = r + 8*hi)
  const int Mrow = M0 + 16 * w;
  const float qscale = 0.03125f; // C^-0.5 folded into q
#pragma unroll
  for (int j = 0; j < 4; ++j)
#pragma unroll
    for (int r = 0; r < 8; ++r) {
      const size_t ridx = (size_t)(Mrow + 8 * hi + r) * HDIM + 16 * j + n;
      qf[ridx] = (_Float16)(acc[0][j][r] * qscale);
      kf[ridx] = (_Float16)acc[1][j][r];
    }
  {
    const int bb = Mrow >> 11;
    const int t  = Mrow & (TDIM - 1);
#pragma unroll
    for (int j = 0; j < 4; ++j) {
      _Float16* vp = vt + ((size_t)(bb * HDIM + 16 * j + n)) * TDIM + t + 8 * hi;
#pragma unroll
      for (int r = 0; r < 8; r += 2)
        *(unsigned*)(vp + r) = pkh2(acc[2][j][r], acc[2][j][r + 1]);
    }
  }
}

// ---------------------------------------------------------------------------
// Kernel 2: flash attention. One wave per 16-query tile.
// S^T = K * Q^T (softmax over keys is lane-local + shfl_xor(16)),
// O^T += V^T * P^T. grid = B*T/16/8 blocks of 256 threads (8 waves).
// ---------------------------------------------------------------------------
__global__ void __launch_bounds__(256)
flash_attn_kernel(const _Float16* __restrict__ qf,
                  const _Float16* __restrict__ kf,
                  const _Float16* __restrict__ vt,
                  float* __restrict__ out) {
  const int tid  = threadIdx.x;
  const int w    = tid >> 5;
  const int lane = tid & 31;
  const int n    = lane & 15;
  const int hi   = lane >> 4;

  const int tile = blockIdx.x * 8 + w;          // 0 .. 1023
  const int b    = tile >> 7;                   // T/16 = 128 tiles per batch
  const int t0   = (tile & 127) * 16;
  const int bT   = b * TDIM;
  const int q_idx = t0 + n;

  // Q^T B-fragments: lane col n, K(h) = 32j + 16hi + h (contiguous 32B)
  v16h qb[2];
#pragma unroll
  for (int j = 0; j < 2; ++j)
    qb[j] = *(const v16h*)(qf + ((size_t)(bT + t0 + n)) * HDIM + 32 * j + 16 * hi);

  v8f oacc[4] = {};
  float m_run = -3.0e30f;
  float l_run = 0.0f;

  const int kend = ((t0 + 16 + 31) >> 5) << 5;  // causal key coverage

  for (int kb = 0; kb < kend; kb += 32) {
    if (kb + 32 < kend) {
      __builtin_prefetch(kf + ((size_t)(bT + kb + 32 + lane)) * HDIM, 0, 3);
      __builtin_prefetch(vt + ((size_t)(b * HDIM + 2 * lane)) * TDIM + kb + 32, 0, 3);
    }

    // ---- S^T: two 16-key tiles, 2 WMMAs each over the head dim ----
    v8f s[2];
#pragma unroll
    for (int ts = 0; ts < 2; ++ts) {
      v8f c = {};
#pragma unroll
      for (int j = 0; j < 2; ++j) {
        const _Float16* kp =
            kf + ((size_t)(bT + kb + 16 * ts + n)) * HDIM + 32 * j;
        v16h a;
        v8h lo = *(const v8h*)(kp + 16 * hi);
        v8h hh = *(const v8h*)(kp + 16 * hi + 8);
#pragma unroll
        for (int i = 0; i < 8; ++i) { a[i] = lo[i]; a[8 + i] = hh[i]; }
        c = __builtin_amdgcn_wmma_f32_16x16x32_f16(
            false, a, false, qb[j], (short)0, c, false, false);
      }
      s[ts] = c;
    }

    // ---- causal mask + block max (key = kb + 16ts + 8hi + r) ----
    float bm = -3.0e30f;
#pragma unroll
    for (int ts = 0; ts < 2; ++ts)
#pragma unroll
      for (int r = 0; r < 8; ++r) {
        const int key = kb + 16 * ts + 8 * hi + r;
        const float sv = (key <= q_idx) ? s[ts][r] : -3.0e30f;
        s[ts][r] = sv;
        bm = fmaxf(bm, sv);
      }
    bm = fmaxf(bm, __shfl_xor(bm, 16, 32));
    const float mnew  = fmaxf(m_run, bm);
    const float alpha = __expf(m_run - mnew);
    m_run = mnew;

    // ---- probabilities, packed to f16 pairs ----
    float rs = 0.0f;
    unsigned ph[2][4];
#pragma unroll
    for (int ts = 0; ts < 2; ++ts)
#pragma unroll
      for (int q2 = 0; q2 < 4; ++q2) {
        const float p0 = __expf(s[ts][2 * q2]     - mnew);
        const float p1 = __expf(s[ts][2 * q2 + 1] - mnew);
        rs += p0 + p1;
        ph[ts][q2] = pkh2(p0, p1);
      }
    rs += __shfl_xor(rs, 16, 32);
    l_run = l_run * alpha + rs;

    // ---- rebuild P^T into B layout (32x16) with cross-lane shuffles ----
    v16h pb;
#pragma unroll
    for (int q2 = 0; q2 < 4; ++q2) {
      const unsigned a0 = (unsigned)__shfl((int)ph[0][q2], n, 32);
      const unsigned b0 = (unsigned)__shfl((int)ph[0][q2], n + 16, 32);
      const unsigned a1 = (unsigned)__shfl((int)ph[1][q2], n, 32);
      const unsigned b1 = (unsigned)__shfl((int)ph[1][q2], n + 16, 32);
      union { unsigned u; _Float16 h[2]; } zl, zh;
      zl.u = hi ? a1 : a0;
      zh.u = hi ? b1 : b0;
      pb[2 * q2]     = zl.h[0]; pb[2 * q2 + 1]     = zl.h[1];
      pb[8 + 2 * q2] = zh.h[0]; pb[8 + 2 * q2 + 1] = zh.h[1];
    }

    // ---- O^T rescale + accumulate V^T * P^T (4 head groups) ----
#pragma unroll
    for (int g = 0; g < 4; ++g) {
#pragma unroll
      for (int r = 0; r < 8; ++r) oacc[g][r] *= alpha;
      const _Float16* vp = vt + ((size_t)(b * HDIM + 16 * g + n)) * TDIM + kb;
      v16h va;
      v8h lo = *(const v8h*)(vp + 16 * hi);
      v8h hh = *(const v8h*)(vp + 16 * hi + 8);
#pragma unroll
      for (int i = 0; i < 8; ++i) { va[i] = lo[i]; va[8 + i] = hh[i]; }
      oacc[g] = __builtin_amdgcn_wmma_f32_16x16x32_f16(
          false, va, false, pb, (short)0, oacc[g], false, false);
    }
  }

  // ---- epilogue: out[b][t0+n][head], head = 16g + 8hi + r ----
  const float inv_l = 1.0f / l_run;
#pragma unroll
  for (int g = 0; g < 4; ++g) {
    float* op = out + ((size_t)(bT + t0 + n)) * HDIM + 16 * g + 8 * hi;
#pragma unroll
    for (int r = 0; r < 8; r += 2) {
      float2 v2 = make_float2(oacc[g][r] * inv_l, oacc[g][r + 1] * inv_l);
      *(float2*)(op + r) = v2;
    }
  }
}

// ---------------------------------------------------------------------------
extern "C" void kernel_launch(void* const* d_in, const int* in_sizes, int n_in,
                              void* d_out, int out_size, void* d_ws, size_t ws_size,
                              hipStream_t stream) {
  const float* x  = (const float*)d_in[0];
  const float* Wk = (const float*)d_in[1];
  const float* Wq = (const float*)d_in[2];
  const float* Wv = (const float*)d_in[3];
  float* out = (float*)d_out;

  // workspace: q | k | v^T (1M halves each) | Wt (3*64*1024 halves) ~ 6.4 MB
  _Float16* qf = (_Float16*)d_ws;
  _Float16* kf = qf + (size_t)BDIM * TDIM * HDIM;
  _Float16* vt = kf + (size_t)BDIM * TDIM * HDIM;
  _Float16* Wt = vt + (size_t)BDIM * TDIM * HDIM;

  dim3 wgrid(16, 3);
  wt_kernel<<<wgrid, 256, 0, stream>>>(Wq, Wk, Wv, Wt);

  proj_qkv_kernel<<<dim3(BDIM * TDIM / 128), 256, 0, stream>>>(x, Wt, qf, kf, vt);

  flash_attn_kernel<<<dim3(BDIM * TDIM / 16 / 8), 256, 0, stream>>>(qf, kf, vt, out);
}
